// MSFN_51694226375100
// MI455X (gfx1250) — matrix operand order repaired and checked by reading, not compiled, not run on hardware
//
#include <hip/hip_runtime.h>

// ---------------------------------------------------------------------------
// MSFN: channel-LN -> (conv3x3 + GELU) x3 -> conv3x3 -> sigmoid -> spatial
// einsum with the ORIGINAL input.  B=32, C=384, H=W=56 (S=3136), N=96.
// All GEMM work on v_wmma_f32_16x16x32_bf16 (wave32, CDNA5).
// ---------------------------------------------------------------------------

typedef __attribute__((ext_vector_type(16))) __bf16 v16bf;
typedef __attribute__((ext_vector_type(8)))  __bf16 v8bf;
typedef __attribute__((ext_vector_type(8)))  float  v8f;

#define BB   32
#define CC   384
#define NN   96
#define HW   56
#define SS   3136   // 56*56
#define HALO 57     // max |flat-s shift| for a 3x3 tap = 56+1
#define STILE 64    // spatial columns per conv block
#define SREG (STILE + 2*HALO)   // 178 staged flat-s positions
#define LDST 48     // padded LDS row stride (halves) -> 96B rows, 32B-aligned frags

// round-to-nearest-even float -> bf16 without relying on fptrunc legality
static __device__ __forceinline__ __bf16 f2bf(float f) {
  unsigned u = __builtin_bit_cast(unsigned, f);
  u += 0x7FFFu + ((u >> 16) & 1u);
  unsigned short h = (unsigned short)(u >> 16);
  return __builtin_bit_cast(__bf16, h);
}

// ---------------------------------------------------------------------------
// 1) Reorder OIHW f32 weights into CDNA5 A-fragment order (16x32 bf16 tiles).
//    Index: (((tap*KC + kc)*6 + rowtile)*32 + lane)*16 + half
//    A 16x32 bf16 lane layout used here (contiguous k):
//      lane<16 : M=lane,     halves = K {0..15}
//      lane>=16: M=lane-16,  halves = K {16..31}
// ---------------------------------------------------------------------------
__global__ void reorder_w_kernel(const float* __restrict__ w,
                                 __bf16* __restrict__ wr,
                                 int CIN, int KC, int total) {
  int e = blockIdx.x * 256 + threadIdx.x;
  if (e >= total) return;
  int i    = e & 15;
  int lane = (e >> 4) & 31;
  int rest = e >> 9;
  int wti  = rest % 6;
  int t2   = rest / 6;
  int kc   = t2 % KC;
  int tap  = t2 / KC;
  int k    = (lane < 16) ? i : i + 16;
  int cin  = kc * 32 + k;
  int cout = wti * 16 + (lane & 15);
  int kh   = tap / 3, kw = tap % 3;
  wr[e] = f2bf(w[((size_t)(cout * CIN + cin) * 3 + kh) * 3 + kw]);
}

// ---------------------------------------------------------------------------
// 2) Channel LayerNorm (biased var, eps=1e-6), coalesced along spatial.
// ---------------------------------------------------------------------------
__global__ void ln_kernel(const float* __restrict__ x,
                          const float* __restrict__ g,
                          const float* __restrict__ bv,
                          __bf16* __restrict__ xn) {
  int p = blockIdx.x * 256 + threadIdx.x;
  if (p >= BB * SS) return;
  int b = p / SS;
  int s = p - b * SS;
  const float* xp = x + (size_t)b * CC * SS + s;
  float sum = 0.f, sq = 0.f;
  for (int c = 0; c < CC; ++c) {
    float v = xp[(size_t)c * SS];
    sum += v; sq += v * v;
  }
  float mean = sum * (1.f / CC);
  float var  = sq * (1.f / CC) - mean * mean;
  float rstd = rsqrtf(var + 1e-6f);
  __bf16* op = xn + (size_t)b * CC * SS + s;
  for (int c = 0; c < CC; ++c) {
    float v = xp[(size_t)c * SS];
    op[(size_t)c * SS] = f2bf((v - mean) * rstd * g[c] + bv[c]);
  }
}

// ---------------------------------------------------------------------------
// 3) 3x3 SAME conv, implicit GEMM on WMMA bf16, double-buffered LDS staging.
//    Block: 192 thr = 6 waves; wave -> one 16-row cout tile; block -> 64
//    contiguous flat-s columns (4 subtiles of 16, 4 accumulators per wave).
//    Per 32-cin chunk: stage the flat-s halo union [s0-57, s0+63+57] into
//    one LDS buffer while the 9 taps x 4 subtiles (36 WMMAs/wave) compute
//    from the other.  Per-lane validity masks row-wrap/border with a zero
//    fragment.  ACT: 0 = exact GELU, 1 = sigmoid.
// ---------------------------------------------------------------------------
template <int CIN, int KC, int ACT>
__global__ __launch_bounds__(192)
void conv3x3_wmma_kernel(const __bf16* __restrict__ in,
                         const __bf16* __restrict__ wr,
                         __bf16* __restrict__ out) {
  __shared__ alignas(32) __bf16 lds[2][SREG * LDST];   // [buf][staged s][cin]

  const int tid  = threadIdx.x;
  const int lane = tid & 31;
  const int wti  = tid >> 5;            // 0..5 output row tile
  const int b    = blockIdx.y;
  const int s0   = blockIdx.x * STILE;
  const int j    = lane & 15;
  const int koff = (lane < 16) ? 0 : 16;

  // per-lane (h,w) of the 4 owned spatial columns — hoisted out of hot loop
  int hj[4], wj[4];
#pragma unroll
  for (int sub = 0; sub < 4; ++sub) {
    int s = s0 + sub * 16 + j;
    hj[sub] = s / HW;
    wj[sub] = s - hj[sub] * HW;
  }

  auto stage = [&](int kc, __bf16* buf) {
    const __bf16* src = in + ((size_t)b * CIN + kc * 32) * SS;
    for (int idx = tid; idx < SREG * 32; idx += 192) {
      int cl = idx / SREG;
      int sl = idx - cl * SREG;
      int s  = s0 - HALO + sl;
      __bf16 v = __builtin_bit_cast(__bf16, (unsigned short)0);
      if (s >= 0 && s < SS) v = src[(size_t)cl * SS + s];
      buf[sl * LDST + cl] = v;
    }
  };

  v8f acc[4] = {};

  stage(0, lds[0]);
  __syncthreads();

  for (int kc = 0; kc < KC; ++kc) {
    const __bf16* cur = lds[kc & 1];
    if (kc + 1 < KC)                    // overlap next stage with compute
      stage(kc + 1, lds[(kc + 1) & 1]);

    for (int tap = 0; tap < 9; ++tap) {
      const int kh = tap / 3 - 1;
      const int kw = tap % 3 - 1;
      // A fragment: pre-swizzled weights, one contiguous 32B per lane,
      // reused by all 4 spatial subtiles
      v16bf a = *(const v16bf*)(wr +
          ((size_t)((tap * KC + kc) * 6 + wti) * 32 + lane) * 16);
      const int shift = HALO + kh * HW + kw;
#pragma unroll
      for (int sub = 0; sub < 4; ++sub) {
        v16bf bf = {};
        if ((unsigned)(hj[sub] + kh) < (unsigned)HW &&
            (unsigned)(wj[sub] + kw) < (unsigned)HW)
          bf = *(const v16bf*)&cur[(sub * 16 + j + shift) * LDST + koff];
        acc[sub] = __builtin_amdgcn_wmma_f32_16x16x32_bf16(
            false, a, false, bf, (short)0, acc[sub], false, false);
      }
    }
    __syncthreads();
  }

  // Epilogue: C/D layout -> M = r + 8*(lane>=16), N = lane&15
  const int mof = (lane < 16) ? 0 : 8;
#pragma unroll
  for (int sub = 0; sub < 4; ++sub) {
#pragma unroll
    for (int r = 0; r < 8; ++r) {
      float v = acc[sub][r];
      if (ACT == 0) v = 0.5f * v * (1.f + erff(v * 0.70710678118654752f));
      else          v = 1.f / (1.f + __expf(-v));
      int cout = wti * 16 + r + mof;
      out[((size_t)b * NN + cout) * SS + s0 + sub * 16 + j] = f2bf(v);
    }
  }
}

// ---------------------------------------------------------------------------
// 4) out[b,c,n] = (1/S) * sum_s sel[b,n,s] * x[b,c,s]  via WMMA bf16,
//    double-buffered LDS staging (decouples global-load latency from the
//    WMMA stream structurally — RA cannot re-serialize it).
//    Block: 6 waves over a 96-row c slab; wave -> one 16-row c tile, all 6
//    n coltiles as accumulators.  Per 32-s chunk:
//      * A slab: 96x32 f32 x -> bf16, written in plain k order (no permutation)
//        (each aligned 16B dest block = contiguous 8-float source run, so
//        staging is 2x(8-float coalesced load -> 1 b128 LDS store)/thread)
//      * B slab: 96x32 bf16 sel, already fragment-linear -> plain b128 copy
//    Rows padded to 48 halves (96B) for bank-conflict relief.
// ---------------------------------------------------------------------------
__global__ __launch_bounds__(192)
void einsum_wmma_kernel(const float* __restrict__ x,
                        const __bf16* __restrict__ sel,
                        float* __restrict__ out) {
  __shared__ alignas(32) __bf16 la[2][NN * LDST];   // 9216B per buffer
  __shared__ alignas(32) __bf16 lb[2][NN * LDST];

  const int tid  = threadIdx.x;
  const int lane = tid & 31;
  const int wti  = tid >> 5;
  const int b    = blockIdx.y;
  const int rt0  = blockIdx.x * 6;      // block's first c rowtile (of 24)
  const int jn   = lane & 15;
  const int koff = (lane < 16) ? 0 : 16;

  auto stage = [&](int kc, int t) {
    // A slab: rows = c, cols = k; dest oct order {0,8,16,24} halves
    const float* xb = x + ((size_t)b * CC + rt0 * 16) * SS + kc * 32;
    for (int idx = tid; idx < NN * 4; idx += 192) {   // 2 iterations
      int row = idx >> 2, oct = idx & 3;
      const float* p = xb + (size_t)row * SS + oct * 8;
      v8bf v;
#pragma unroll
      for (int i = 0; i < 8; ++i) v[i] = f2bf(p[i]);
      *(v8bf*)&la[t][row * LDST + oct * 8] = v;
    }
    // B slab: fragment-linear already; straight 16B copies
    const __bf16* sb = sel + (size_t)b * NN * SS + kc * 32;
    for (int idx = tid; idx < NN * 4; idx += 192) {   // 2 iterations
      int row = idx >> 2, q = idx & 3;
      *(uint4*)&lb[t][row * LDST + q * 8] =
          *(const uint4*)(sb + (size_t)row * SS + q * 8);
    }
  };

  v8f acc[6] = {};

  stage(0, 0);
  __syncthreads();

  for (int kc = 0; kc < SS / 32; ++kc) {
    const int t = kc & 1;
    if (kc + 1 < SS / 32)               // overlap next stage with compute
      stage(kc + 1, t ^ 1);

    v16bf a = *(const v16bf*)&la[t][(wti * 16 + jn) * LDST + koff];
#pragma unroll
    for (int nt = 0; nt < 6; ++nt) {
      v16bf bf = *(const v16bf*)&lb[t][(nt * 16 + jn) * LDST + koff];
      acc[nt] = __builtin_amdgcn_wmma_f32_16x16x32_bf16(
          false, a, false, bf, (short)0, acc[nt], false, false);
    }
    __syncthreads();
  }

  const int mof = (lane < 16) ? 0 : 8;
#pragma unroll
  for (int nt = 0; nt < 6; ++nt) {
#pragma unroll
    for (int r = 0; r < 8; ++r) {
      int cr = (rt0 + wti) * 16 + r + mof;
      out[((size_t)b * CC + cr) * NN + nt * 16 + jn] =
          acc[nt][r] * (1.f / (float)SS);
    }
  }
}

// ---------------------------------------------------------------------------
// Workspace layout (bytes):
//   xn  : 0                       77,070,336  (B*C*S bf16)
//   hA  : 77,070,336              19,267,584  (B*N*S bf16)
//   hB  : 96,337,920              19,267,584
//   sel : 115,605,504             19,267,584
//   wr1 : 134,873,088                663,552  (9*12*6*512 bf16)
//   wr2 : 135,536,640                165,888
//   wr3 : 135,702,528                165,888
//   wr4 : 135,868,416                165,888   -> total ~136.0 MB
// ---------------------------------------------------------------------------
extern "C" void kernel_launch(void* const* d_in, const int* in_sizes, int n_in,
                              void* d_out, int out_size, void* d_ws, size_t ws_size,
                              hipStream_t stream) {
  const float* x  = (const float*)d_in[0];
  const float* g  = (const float*)d_in[1];
  const float* bv = (const float*)d_in[2];
  const float* w1 = (const float*)d_in[3];
  const float* w2 = (const float*)d_in[4];
  const float* w3 = (const float*)d_in[5];
  const float* w4 = (const float*)d_in[6];

  char* ws = (char*)d_ws;
  __bf16* xn  = (__bf16*)(ws + 0);
  __bf16* hA  = (__bf16*)(ws + 77070336ull);
  __bf16* hB  = (__bf16*)(ws + 96337920ull);
  __bf16* sel = (__bf16*)(ws + 115605504ull);
  __bf16* wr1 = (__bf16*)(ws + 134873088ull);
  __bf16* wr2 = (__bf16*)(ws + 135536640ull);
  __bf16* wr3 = (__bf16*)(ws + 135702528ull);
  __bf16* wr4 = (__bf16*)(ws + 135868416ull);

  // weight pre-swizzle into WMMA A-fragment order
  {
    int t1 = 9 * 12 * 6 * 512;   // Cin=384
    int t2 = 9 * 3 * 6 * 512;    // Cin=96
    reorder_w_kernel<<<(t1 + 255) / 256, 256, 0, stream>>>(w1, wr1, CC, 12, t1);
    reorder_w_kernel<<<(t2 + 255) / 256, 256, 0, stream>>>(w2, wr2, NN, 3, t2);
    reorder_w_kernel<<<(t2 + 255) / 256, 256, 0, stream>>>(w3, wr3, NN, 3, t2);
    reorder_w_kernel<<<(t2 + 255) / 256, 256, 0, stream>>>(w4, wr4, NN, 3, t2);
  }

  // channel LayerNorm -> bf16 activations
  ln_kernel<<<(BB * SS + 255) / 256, 256, 0, stream>>>(x, g, bv, xn);

  // conv chain: 49 spatial tiles x 32 batches, 6 waves per block
  dim3 cgrid(SS / STILE, BB);
  conv3x3_wmma_kernel<CC, 12, 0><<<cgrid, 192, 0, stream>>>(xn, wr1, hA);
  conv3x3_wmma_kernel<NN, 3,  0><<<cgrid, 192, 0, stream>>>(hA, wr2, hB);
  conv3x3_wmma_kernel<NN, 3,  0><<<cgrid, 192, 0, stream>>>(hB, wr3, hA);
  conv3x3_wmma_kernel<NN, 3,  1><<<cgrid, 192, 0, stream>>>(hA, wr4, sel);

  // final projection against the ORIGINAL (pre-LN) input
  dim3 egrid(CC / (16 * 6), BB);   // 4 x 32 blocks, 24 rowtiles / 6 waves
  einsum_wmma_kernel<<<egrid, 192, 0, stream>>>(x, sel, (float*)d_out);
}
